// Net_15255723836145
// MI455X (gfx1250) — matrix operand order repaired and checked by reading, not compiled, not run on hardware
//
#include <hip/hip_runtime.h>
#include <hip/hip_bf16.h>
#include <cstdint>

// ---------------- problem constants ----------------
#define B_SZ 512
#define NSUB 70
#define TLEN 128
#define INPD 4
#define NCLS 5
#define XFEAT 285                      // 2*NSUB + 2*NSUB + NCLS
#define XCOLS 288                      // padded to multiple of 32/4
#define SEQ_PER_RNN (B_SZ * NSUB)      // 35840
#define SEQ_FLOATS  (TLEN * INPD)      // 512 floats per sequence
#define THREADS 128                    // sequences per block
#define BLOCKS_PER_RNN (SEQ_PER_RNN / THREADS)  // 280
#define CT 8                           // timesteps per chunk
#define CHUNK_FLOATS (CT * INPD)       // 32 floats / seq / chunk
#define NCHUNK (TLEN / CT)             // 16 chunks
#define LROW 36                        // padded LDS row stride (floats): 32 data + 4 pad
#define BUF_FLOATS (THREADS * LROW)    // 4608 floats per buffer (18432 B)
#define X_FLOATS ((size_t)B_SZ * XCOLS)      // X matrix in ws
#define WPAD_FLOATS (16 * XCOLS)             // zero-padded fc2_W tile in ws

typedef unsigned int v4u   __attribute__((ext_vector_type(4)));
typedef int          v8i_t __attribute__((ext_vector_type(8)));
typedef int          v4i_t __attribute__((ext_vector_type(4)));
typedef float        v2f_t __attribute__((ext_vector_type(2)));
typedef float        v8f_t __attribute__((ext_vector_type(8)));
typedef _Float16     v16h_t __attribute__((ext_vector_type(16)));

#ifndef __has_builtin
#define __has_builtin(x) 0
#endif

#if __has_builtin(__builtin_amdgcn_tensor_load_to_lds)
#define HAVE_TDM 1
#else
#define HAVE_TDM 0
#endif

#if __has_builtin(__builtin_amdgcn_wmma_f32_16x16x4_f32)
#define HAVE_WMMA_F32X4 1
#else
#define HAVE_WMMA_F32X4 0
#endif

// Native v_tanh_f32 if available; else 4-op exp/rcp approx (never libm tanhf:
// ~20 VALU ops/step would push the kernel past the ~1 op/byte bandwidth budget).
#if __has_builtin(__builtin_amdgcn_tanhf)
#define FTANH(a) __builtin_amdgcn_tanhf(a)
#elif __has_builtin(__builtin_amdgcn_tanh_f32)
#define FTANH(a) __builtin_amdgcn_tanh_f32(a)
#else
#define FTANH(a) (1.0f - 2.0f * __builtin_amdgcn_rcpf(__expf(2.0f * (a)) + 1.0f))
#endif

// ---------------- TDM issue: 2D tile (THREADS rows x CHUNK_FLOATS f32),
// row stride SEQ_FLOATS, LDS-padded 4 DWORDs every 32 DWORDs ----------------
#if HAVE_TDM
__device__ __forceinline__ void tdm_issue(unsigned int lds_byte_addr, const float* gsrc) {
  unsigned long long ga = (unsigned long long)(size_t)gsrc;
  v4u g0;
  g0.x = 1u;                                            // count=1, is_restore=0
  g0.y = lds_byte_addr;                                 // [63:32] lds_addr
  g0.z = (unsigned int)(ga & 0xFFFFFFFFull);            // global_addr[31:0]
  g0.w = (unsigned int)((ga >> 32) & 0x1FFFFFFu)        // global_addr[56:32]
       | (2u << 30);                                    // type=2 ("image")
  v8i_t g1;
  g1[0] = (2 << 16)     // data_size = 4B
        | (1 << 20)     // pad_enable
        | (4 << 22)     // pad_interval: 32 DWORDs
        | (3 << 25);    // pad_amount: 4 DWORDs
  g1[1] = (int)(((unsigned)SEQ_FLOATS & 0xFFFFu) << 16);                // tensor_dim0 lo16
  g1[2] = (int)(((unsigned)SEQ_FLOATS >> 16)
        | (((unsigned)THREADS & 0xFFFFu) << 16));                       // dim0 hi / dim1 lo
  g1[3] = (int)(((unsigned)THREADS >> 16)
        | ((unsigned)CHUNK_FLOATS << 16));                              // dim1 hi / tile_dim0
  g1[4] = (int)THREADS;                                                 // tile_dim1 (tile_dim2=0)
  g1[5] = SEQ_FLOATS;                                                   // tensor_dim0_stride lo32
  g1[6] = 0;
  g1[7] = 0;
  v4i_t g2 = {0, 0, 0, 0};
  v4i_t g3 = {0, 0, 0, 0};
#if __clang_major__ >= 23
  v8i_t g4 = {0, 0, 0, 0, 0, 0, 0, 0};
  __builtin_amdgcn_tensor_load_to_lds(g0, g1, g2, g3, g4, 0);
#else
  __builtin_amdgcn_tensor_load_to_lds(g0, g1, g2, g3, 0);
#endif
}
#endif

// ---------------- Kernel 0a: features + zero padding into X[:,280:288] ------
__global__ void fill_features_kernel(const float* __restrict__ feat, float* __restrict__ X) {
  const int b = blockIdx.x;
  const int j = threadIdx.x;  // 0..7
  X[(size_t)b * XCOLS + 280 + j] = (j < NCLS) ? feat[b * NCLS + j] : 0.0f;
}

// ---------------- Kernel 0b: zero-pad fc2_W into a 16x288 tile ---------------
// Wpad[n][k] = fc2_W[n][k] for n<5, k<285; else 0. Makes the WMMA loop
// branch-free (EXEC stays all-ones, loads are plain b64).
__global__ void pad_w_kernel(const float* __restrict__ W, float* __restrict__ Wpad) {
  const int n = blockIdx.x;    // 0..15
  const int k = threadIdx.x;   // 0..287
  Wpad[n * XCOLS + k] = (n < NCLS && k < XFEAT) ? W[n * XFEAT + k] : 0.0f;
}

// ---------------- Kernel 1: batched tanh-RNN, TDM double-buffered ------------
__global__ __launch_bounds__(THREADS) void rnn_kernel(
    const float* __restrict__ seek, const float* __restrict__ svc,
    const float* __restrict__ WihS, const float* __restrict__ WhhS,
    const float* __restrict__ bihS, const float* __restrict__ bhhS,
    const float* __restrict__ WihV, const float* __restrict__ WhhV,
    const float* __restrict__ bihV, const float* __restrict__ bhhV,
    float* __restrict__ X) {
  __shared__ float lbuf[2 * BUF_FLOATS];
  const int tid = threadIdx.x;
  const bool isv = (blockIdx.x >= BLOCKS_PER_RNN);
  const int blk = isv ? (int)blockIdx.x - BLOCKS_PER_RNN : (int)blockIdx.x;
  const int q0 = blk * THREADS;

  const float* src = isv ? svc  : seek;
  const float* Wih = isv ? WihV : WihS;
  const float* Whh = isv ? WhhV : WhhS;
  const float* bih = isv ? bihV : bihS;
  const float* bhh = isv ? bhhV : bhhS;

  const float w00 = Wih[0], w01 = Wih[1], w02 = Wih[2], w03 = Wih[3];
  const float w10 = Wih[4], w11 = Wih[5], w12 = Wih[6], w13 = Wih[7];
  const float u00 = Whh[0], u01 = Whh[1], u10 = Whh[2], u11 = Whh[3];
  const float b0 = bih[0] + bhh[0], b1 = bih[1] + bhh[1];

  const float* gseq = src + (size_t)q0 * SEQ_FLOATS;

#if HAVE_TDM
  const unsigned int lds0 = (unsigned int)(size_t)&lbuf[0];
  if (tid < 32) tdm_issue(lds0, gseq);   // prologue: chunk 0 -> buffer 0
#endif

  float h0 = 0.0f, h1 = 0.0f;
  for (int c = 0; c < NCHUNK; ++c) {
    const int cur = c & 1;
#if HAVE_TDM
    if (tid < 32) {
      if (c + 1 < NCHUNK) {
        tdm_issue(lds0 + (unsigned int)((cur ^ 1) * BUF_FLOATS * 4),
                  gseq + (size_t)(c + 1) * CHUNK_FLOATS);
        __builtin_amdgcn_s_wait_tensorcnt(1);   // chunk c complete (in-order)
      } else {
        __builtin_amdgcn_s_wait_tensorcnt(0);
      }
    }
    __syncthreads();
#else
    // fallback: cooperative synchronous staging, same padded layout
    for (int i = 0; i < 8; ++i) {
      const int l = i * THREADS + tid;      // 1024 float4 per chunk
      const int s = l >> 3;
      const int j = l & 7;
      *(float4*)&lbuf[cur * BUF_FLOATS + s * LROW + j * 4] =
          *(const float4*)(gseq + (size_t)s * SEQ_FLOATS + c * CHUNK_FLOATS + j * 4);
    }
    __syncthreads();
#endif
    const float* xrow = &lbuf[cur * BUF_FLOATS + tid * LROW];
#pragma unroll
    for (int t = 0; t < CT; ++t) {
      const float4 x = *(const float4*)(xrow + t * 4);
      float a0 = fmaf(w00, x.x, fmaf(w01, x.y, fmaf(w02, x.z,
                 fmaf(w03, x.w, fmaf(u00, h0, fmaf(u01, h1, b0))))));
      float a1 = fmaf(w10, x.x, fmaf(w11, x.y, fmaf(w12, x.z,
                 fmaf(w13, x.w, fmaf(u10, h0, fmaf(u11, h1, b1))))));
      h0 = FTANH(a0);
      h1 = FTANH(a1);
    }
    __syncthreads();  // protect buffer `cur` before it is refilled at c+2
  }

  const int q  = q0 + tid;
  const int bb = q / NSUB;
  const int ss = q - bb * NSUB;
  const int col = (isv ? 2 * NSUB : 0) + 2 * ss;
  X[(size_t)bb * XCOLS + col]     = h0;
  X[(size_t)bb * XCOLS + col + 1] = h1;
}

// ---------------- Kernel 2: logits = X @ Wpad^T + b via WMMA -----------------
// Branch-free: both operands zero-padded to K=288, N=16.
__global__ __launch_bounds__(32) void gemm_kernel(const float* __restrict__ X,
                                                  const float* __restrict__ Wpad,
                                                  const float* __restrict__ bias,
                                                  float* __restrict__ out) {
  const int lane = threadIdx.x;
  const int m0   = blockIdx.x * 16;
  const int mrow = lane & 15;
  const int hi   = lane >> 4;
  const int ncol = lane & 15;
  v8f_t acc = {0.f, 0.f, 0.f, 0.f, 0.f, 0.f, 0.f, 0.f};
#if HAVE_WMMA_F32X4
  // A 16x4 f32: lanes 0-15 K=k..k+1, lanes 16-31 K=k+2..k+3; B mirrored.
  const float* arow = X    + (size_t)(m0 + mrow) * XCOLS + hi * 2;
  const float* brow = Wpad + (size_t)ncol * XCOLS        + hi * 2;
#pragma unroll 4
  for (int k = 0; k < XCOLS; k += 4) {
    const v2f_t a = *(const v2f_t*)(arow + k);
    const v2f_t b = *(const v2f_t*)(brow + k);
    acc = __builtin_amdgcn_wmma_f32_16x16x4_f32(false, a, false, b,
                                                (short)0, acc, false, false);
  }
#else
  // fallback: codegen-confirmed f16 WMMA (K=32), operands converted on the fly.
  for (int kb = 0; kb < XCOLS; kb += 32) {
    v16h_t a, b;
#pragma unroll
    for (int e = 0; e < 16; ++e) {
      const int ka = kb + (e < 8 ? e : e + 8) + hi * 8;
      a[e] = (_Float16)X[(size_t)(m0 + mrow) * XCOLS + ka];
      const int kw = kb + (e < 8 ? e : e + 8) + hi * 8;
      b[e] = (_Float16)Wpad[(size_t)ncol * XCOLS + kw];
    }
    acc = __builtin_amdgcn_wmma_f32_16x16x32_f16(false, a, false, b,
                                                 (short)0, acc, false, false);
  }
#endif
  if (ncol < NCLS) {
    const float bv = bias[ncol];
#pragma unroll
    for (int r = 0; r < 8; ++r) {
      const int row = hi ? r + 8 : r;   // lanes 0-15: M=r; lanes 16-31: M=r+8
      out[(m0 + row) * NCLS + ncol] = acc[r] + bv;
    }
  }
}

// ---------------- Kernel 3: softmax over axis=0 (batch), in place ------------
__global__ __launch_bounds__(B_SZ) void softmax_kernel(float* __restrict__ out) {
  __shared__ float red[B_SZ];
  const int n = blockIdx.x;     // class
  const int t = threadIdx.x;    // batch row
  const float v = out[t * NCLS + n];
  red[t] = v;
  __syncthreads();
  for (int s = B_SZ / 2; s > 0; s >>= 1) {
    if (t < s) red[t] = fmaxf(red[t], red[t + s]);
    __syncthreads();
  }
  const float mx = red[0];
  __syncthreads();
  const float e = __expf(v - mx);
  red[t] = e;
  __syncthreads();
  for (int s = B_SZ / 2; s > 0; s >>= 1) {
    if (t < s) red[t] += red[t + s];
    __syncthreads();
  }
  out[t * NCLS + n] = e / red[0];
}

// ---------------- launch -----------------------------------------------------
extern "C" void kernel_launch(void* const* d_in, const int* in_sizes, int n_in,
                              void* d_out, int out_size, void* d_ws, size_t ws_size,
                              hipStream_t stream) {
  const float* seek = (const float*)d_in[0];
  const float* svc  = (const float*)d_in[1];
  const float* feat = (const float*)d_in[2];
  const float* WihS = (const float*)d_in[3];
  const float* WhhS = (const float*)d_in[4];
  const float* bihS = (const float*)d_in[5];
  const float* bhhS = (const float*)d_in[6];
  const float* WihV = (const float*)d_in[7];
  const float* WhhV = (const float*)d_in[8];
  const float* bihV = (const float*)d_in[9];
  const float* bhhV = (const float*)d_in[10];
  const float* fc2W = (const float*)d_in[11];
  const float* fc2b = (const float*)d_in[12];
  float* X    = (float*)d_ws;             // [512, 288] f32
  float* Wpad = X + X_FLOATS;             // [16, 288] f32, zero-padded fc2_W
  float* out  = (float*)d_out;            // [512, 5] f32

  fill_features_kernel<<<dim3(B_SZ), dim3(8), 0, stream>>>(feat, X);
  pad_w_kernel<<<dim3(16), dim3(XCOLS), 0, stream>>>(fc2W, Wpad);
  rnn_kernel<<<dim3(2 * BLOCKS_PER_RNN), dim3(THREADS), 0, stream>>>(
      seek, svc, WihS, WhhS, bihS, bhhS, WihV, WhhV, bihV, bhhV, X);
  gemm_kernel<<<dim3(B_SZ / 16), dim3(32), 0, stream>>>(X, Wpad, fc2b, out);
  softmax_kernel<<<dim3(NCLS), dim3(B_SZ), 0, stream>>>(out);

  (void)in_sizes; (void)n_in; (void)out_size; (void)ws_size;
}
